// OptimizedMultiHeadAttention_53661321397042
// MI455X (gfx1250) — matrix operand order repaired and checked by reading, not compiled, not run on hardware
//
#include <hip/hip_runtime.h>

// ---------------------------------------------------------------------------
// MHA forward for MI455X (gfx1250, wave32, WMMA, TDM).
//   1) convert x / w_qkv / wo fp32 -> bf16 (workspace)
//   2) qkv = x @ w_qkv^T           (bf16 WMMA GEMM, 128x128 block tile)
//   3) flash attention per (b,h)   (64-key blocks, TDM K-tile load, exp2 softmax)
//   4) out = y @ wo^T              (bf16 WMMA GEMM, fp32 out)
// ---------------------------------------------------------------------------

typedef unsigned short u16;
typedef unsigned int   u32;
typedef __attribute__((ext_vector_type(8)))  u16    u16x8;
typedef __attribute__((ext_vector_type(16))) u16    u16x16;
typedef __attribute__((ext_vector_type(16))) __bf16 v16bf;
typedef __attribute__((ext_vector_type(8)))  float  v8f;
typedef __attribute__((ext_vector_type(4)))  u32    uint32x4;
typedef __attribute__((ext_vector_type(8)))  int    int32x8;
typedef __attribute__((ext_vector_type(4)))  int    int32x4;

union frag16 { u16x16 v; u16x8 h[2]; };

#define BATCH 4
#define TSEQ  2048
#define HEADS 16
#define DHEAD 64
#define EMB   1024
#define MT    (BATCH * TSEQ)      /* 8192 token rows */
#define QKVW  (3 * HEADS * DHEAD) /* 3072 */

#if defined(__has_builtin)
#  if __has_builtin(__builtin_amdgcn_tensor_load_to_lds) && \
      __has_builtin(__builtin_amdgcn_s_wait_tensorcnt)
#    define HAVE_TDM 1
#  endif
#endif
#ifndef HAVE_TDM
#  define HAVE_TDM 0
#endif

__device__ __forceinline__ u16 f32_to_bf16(float f) {
  u32 u = __builtin_bit_cast(u32, f);
  u += 0x7fffu + ((u >> 16) & 1u);       // round-to-nearest-even
  return (u16)(u >> 16);
}

__device__ __forceinline__ v8f wmma_bf16(const frag16& a, const frag16& b, v8f c) {
  return __builtin_amdgcn_wmma_f32_16x16x32_bf16(
      false, __builtin_bit_cast(v16bf, a.v),
      false, __builtin_bit_cast(v16bf, b.v),
      (short)0, c, false, false);
}

// ---------------------------------------------------------------------------
// fp32 -> bf16 elementwise convert
// ---------------------------------------------------------------------------
__global__ void cvt_f32_bf16(const float* __restrict__ src, u16* __restrict__ dst, int n) {
  int i = blockIdx.x * blockDim.x + threadIdx.x;
  int stride = gridDim.x * blockDim.x;
  for (; i < n; i += stride) dst[i] = f32_to_bf16(src[i]);
}

// ---------------------------------------------------------------------------
// GEMM: C[M,N] = A[M,K](bf16) * W[N,K]^T(bf16).
// Block tile 128x128: 8 waves as 4(M) x 2(N); each wave owns 32x64
// (2 A-frags x 4 B-frags -> 8 WMMA per K-step of 32).
// ---------------------------------------------------------------------------
template <int OUT_F32>
__global__ void __launch_bounds__(256)
gemm_bf16_wmma(const u16* __restrict__ A, const u16* __restrict__ W,
               void* __restrict__ Cout, int M, int N, int K) {
  const int lane = threadIdx.x & 31;
  const int wv   = threadIdx.x >> 5;
  const int wm   = wv >> 1;                 // 0..3
  const int wn   = wv & 1;                  // 0..1
  const int m16  = lane & 15;
  const int hf   = lane >> 4;               // wave32 half
  const int rowA = blockIdx.x * 128 + wm * 32;
  const int nco  = blockIdx.y * 128 + wn * 64;

  const u16* arow0 = A + (size_t)(rowA + m16) * K;
  const u16* arow1 = A + (size_t)(rowA + 16 + m16) * K;
  v8f acc[2][4] = {};

  for (int k0 = 0; k0 < K; k0 += 32) {
    __builtin_prefetch(arow0 + k0 + 128, 0, 0);   // global_prefetch_b8
    frag16 af[2];
    af[0].h[0] = *(const u16x8*)(arow0 + k0 + 8 * hf);
    af[0].h[1] = *(const u16x8*)(arow0 + k0 + 16 + 8 * hf);
    af[1].h[0] = *(const u16x8*)(arow1 + k0 + 8 * hf);
    af[1].h[1] = *(const u16x8*)(arow1 + k0 + 16 + 8 * hf);
#pragma unroll
    for (int t = 0; t < 4; ++t) {
      const u16* wrow = W + (size_t)(nco + t * 16 + m16) * K + k0 + 8 * hf;
      frag16 bfr;
      bfr.h[0] = *(const u16x8*)(wrow);
      bfr.h[1] = *(const u16x8*)(wrow + 16);
      acc[0][t] = wmma_bf16(af[0], bfr, acc[0][t]);
      acc[1][t] = wmma_bf16(af[1], bfr, acc[1][t]);
    }
  }

#pragma unroll
  for (int u = 0; u < 2; ++u)
#pragma unroll
    for (int t = 0; t < 4; ++t)
#pragma unroll
      for (int i = 0; i < 8; ++i) {
        const int r = rowA + u * 16 + i + 8 * hf;
        const int c = nco + t * 16 + m16;
        if (OUT_F32)
          ((float*)Cout)[(size_t)r * N + c] = acc[u][t][i];
        else
          ((u16*)Cout)[(size_t)r * N + c] = f32_to_bf16(acc[u][t][i]);
      }
}

// ---------------------------------------------------------------------------
// Flash attention (causal). Block = (128-query tile, head, batch), 8 waves x
// 16 q-rows, 64-key blocks. K-tile DMA'd by the Tensor Data Mover when
// available; V transposed manually into LDS; softmax in exp2 domain.
// ---------------------------------------------------------------------------
__global__ void __launch_bounds__(256)
flash_attn_bf16(const u16* __restrict__ qkv, u16* __restrict__ Y) {
  __shared__ u16 Kt[64 * 64];        // K block row-major [krow][dh]      8 KB
  __shared__ u16 Vt[64 * 64];        // V block transposed [dh][krow]     8 KB
  __shared__ u16 Pt[8 * 16 * 64];    // per-wave P bounce               16 KB

  const int lane = threadIdx.x & 31;
  const int wv   = threadIdx.x >> 5;
  const int m16  = lane & 15;
  const int hf   = lane >> 4;
  const int qb = blockIdx.x, hd = blockIdx.y, bb = blockIdx.z;
  const int qrow0 = qb * 128 + wv * 16;
  const size_t tokbase = (size_t)bb * TSEQ;

  // Q fragments (DH=64 -> two K=32 fragments)
  frag16 qa[2];
  {
    const u16* qp = qkv + (tokbase + qrow0 + m16) * QKVW + hd * DHEAD;
#pragma unroll
    for (int s = 0; s < 2; ++s) {
      qa[s].h[0] = *(const u16x8*)(qp + 32 * s + 8 * hf);
      qa[s].h[1] = *(const u16x8*)(qp + 32 * s + 16 + 8 * hf);
    }
  }

  v8f o[4] = {};
  float rmax[8], rsum[8];
#pragma unroll
  for (int i = 0; i < 8; ++i) { rmax[i] = -1e30f; rsum[i] = 0.0f; }

  const float sc2 = 0.125f * 1.44269504088896f;   // scale * log2(e)
  const int nj = qb * 2 + 2;                      // 64-key blocks, uniform

  for (int j = 0; j < nj; ++j) {
    __syncthreads();                              // protect Kt/Vt reuse
    // ---- V block: manual transposed stage ----
#pragma unroll
    for (int e = 0; e < 2; ++e) {
      const int id = threadIdx.x * 2 + e;         // 0..511
      const int r = id >> 3, c8 = (id & 7) * 8;
      const size_t tok = (tokbase + (size_t)j * 64 + r) * QKVW + hd * DHEAD;
      u16x8 vvv = *(const u16x8*)(qkv + tok + 2 * EMB + c8);
#pragma unroll
      for (int q = 0; q < 8; ++q) Vt[(c8 + q) * 64 + r] = vvv[q];
    }
    // ---- K block: TDM DMA (fallback: vector loads) ----
#if HAVE_TDM
    if (wv == 0) {
      const size_t ga =
          (size_t)(const void*)(qkv + (tokbase + (size_t)j * 64) * QKVW + EMB + hd * DHEAD);
      const u32 la = (u32)(size_t)(void*)Kt;
      uint32x4 g0 = { 1u,                                   // count=1
                      la,                                   // lds_addr
                      (u32)ga,                              // global_addr lo
                      (u32)((ga >> 32) & 0x01FFFFFFu) | (2u << 30) }; // hi | type=2
      int32x8 g1 = { (int)0x00010000,                       // data_size=2B
                     (int)((QKVW & 0xFFFF) << 16),          // tensor_dim0 lo16
                     (int)(((u32)MT & 0xFFFFu) << 16),      // dim0 hi / dim1 lo
                     (int)(64u << 16),                      // dim1 hi / tile_dim0=64
                     (int)64,                               // tile_dim1=64, tile_dim2=0
                     (int)QKVW,                             // tensor_dim0_stride lo32
                     0, 0 };
      int32x4 gz = { 0, 0, 0, 0 };
#if __clang_major__ >= 23
      int32x8 gz8 = { 0, 0, 0, 0, 0, 0, 0, 0 };
      __builtin_amdgcn_tensor_load_to_lds(g0, g1, gz, gz, gz8, 0);
#else
      __builtin_amdgcn_tensor_load_to_lds(g0, g1, gz, gz, 0);
#endif
      __builtin_amdgcn_s_wait_tensorcnt(0);
    }
#else
#pragma unroll
    for (int e = 0; e < 2; ++e) {
      const int id = threadIdx.x * 2 + e;
      const int r = id >> 3, c8 = (id & 7) * 8;
      const size_t tok = (tokbase + (size_t)j * 64 + r) * QKVW + hd * DHEAD;
      *(u16x8*)(Kt + r * 64 + c8) = *(const u16x8*)(qkv + tok + EMB + c8);
    }
#endif
    __syncthreads();

    // ---- S(16x64) = Q x K^T : four 16x16 accumulators ----
    v8f s[4] = {};
#pragma unroll
    for (int ks = 0; ks < 2; ++ks)
#pragma unroll
      for (int t = 0; t < 4; ++t) {
        frag16 kb;
        kb.h[0] = *(const u16x8*)(Kt + (t * 16 + m16) * 64 + ks * 32 + 8 * hf);
        kb.h[1] = *(const u16x8*)(Kt + (t * 16 + m16) * 64 + ks * 32 + 8 * hf + 16);
        s[t] = wmma_bf16(qa[ks], kb, s[t]);
      }

    // ---- causal mask (skipped for interior blocks) + online softmax ----
    float p[4][8];
    if (j * 64 + 63 <= qrow0) {                   // wave-uniform: full block
#pragma unroll
      for (int t = 0; t < 4; ++t)
#pragma unroll
        for (int i = 0; i < 8; ++i) p[t][i] = s[t][i] * sc2;
    } else {
#pragma unroll
      for (int t = 0; t < 4; ++t)
#pragma unroll
        for (int i = 0; i < 8; ++i) {
          const int qr = qrow0 + i + 8 * hf;
          const int kc = j * 64 + t * 16 + m16;
          p[t][i] = (kc <= qr) ? s[t][i] * sc2 : -1e30f;
        }
    }
    float mloc[8];
#pragma unroll
    for (int i = 0; i < 8; ++i)
      mloc[i] = fmaxf(fmaxf(p[0][i], p[1][i]), fmaxf(p[2][i], p[3][i]));
#pragma unroll
    for (int i = 0; i < 8; ++i)
      for (int off = 1; off < 16; off <<= 1)
        mloc[i] = fmaxf(mloc[i], __shfl_xor(mloc[i], off, 32));
    float fsc[8];
#pragma unroll
    for (int i = 0; i < 8; ++i) {
      const float nm = fmaxf(rmax[i], mloc[i]);
      fsc[i] = exp2f(rmax[i] - nm);               // v_exp_f32
      rmax[i] = nm;
      float ls = 0.0f;
#pragma unroll
      for (int t = 0; t < 4; ++t) { p[t][i] = exp2f(p[t][i] - nm); ls += p[t][i]; }
      for (int off = 1; off < 16; off <<= 1) ls += __shfl_xor(ls, off, 32);
      rsum[i] = rsum[i] * fsc[i] + ls;
    }
#pragma unroll
    for (int t = 0; t < 4; ++t)
#pragma unroll
      for (int i = 0; i < 8; ++i) o[t][i] *= fsc[i];

    // ---- P (C-layout) -> LDS bf16 -> two A-fragments (intra-wave) ----
    u16* Pw = Pt + wv * 1024;
#pragma unroll
    for (int t = 0; t < 4; ++t)
#pragma unroll
      for (int i = 0; i < 8; ++i)
        Pw[(i + 8 * hf) * 64 + t * 16 + m16] = f32_to_bf16(p[t][i]);
    asm volatile("s_wait_dscnt 0" ::: "memory");
    frag16 pa[2];
#pragma unroll
    for (int s2 = 0; s2 < 2; ++s2) {
      pa[s2].h[0] = *(const u16x8*)(Pw + m16 * 64 + s2 * 32 + 8 * hf);
      pa[s2].h[1] = *(const u16x8*)(Pw + m16 * 64 + s2 * 32 + 16 + 8 * hf);
    }

    // ---- O += P(16x64) x V(64x64) ----
#pragma unroll
    for (int t = 0; t < 4; ++t)
#pragma unroll
      for (int s2 = 0; s2 < 2; ++s2) {
        frag16 vb;
        vb.h[0] = *(const u16x8*)(Vt + (t * 16 + m16) * 64 + s2 * 32 + 8 * hf);
        vb.h[1] = *(const u16x8*)(Vt + (t * 16 + m16) * 64 + s2 * 32 + 8 * hf + 16);
        o[t] = wmma_bf16(pa[s2], vb, o[t]);
      }
  }

  // ---- epilogue: normalize, write y (bf16) ----
#pragma unroll
  for (int t = 0; t < 4; ++t)
#pragma unroll
    for (int i = 0; i < 8; ++i) {
      const int r = qrow0 + i + 8 * hf;
      const int c = hd * DHEAD + t * 16 + m16;
      Y[(tokbase + r) * EMB + c] = f32_to_bf16(o[t][i] / rsum[i]);
    }
}

// ---------------------------------------------------------------------------
// Launcher
// ---------------------------------------------------------------------------
extern "C" void kernel_launch(void* const* d_in, const int* in_sizes, int n_in,
                              void* d_out, int out_size, void* d_ws, size_t ws_size,
                              hipStream_t stream) {
  const float* x     = (const float*)d_in[0];
  const float* w_qkv = (const float*)d_in[1];
  const float* wo    = (const float*)d_in[2];
  float* out = (float*)d_out;

  char* ws = (char*)d_ws;
  size_t off = 0;
  auto alloc = [&](size_t bytes) -> void* {
    void* p = ws + off;
    off += (bytes + 255) & ~(size_t)255;
    return p;
  };
  u16* xb    = (u16*)alloc((size_t)MT * EMB * 2);
  u16* wqkvb = (u16*)alloc((size_t)QKVW * EMB * 2);
  u16* wob   = (u16*)alloc((size_t)EMB * EMB * 2);
  u16* qkvb  = (u16*)alloc((size_t)MT * QKVW * 2);
  u16* yb    = (u16*)alloc((size_t)MT * EMB * 2);
  (void)ws_size;

  cvt_f32_bf16<<<2048, 256, 0, stream>>>(x, xb, MT * EMB);
  cvt_f32_bf16<<<2048, 256, 0, stream>>>(w_qkv, wqkvb, QKVW * EMB);
  cvt_f32_bf16<<<1024, 256, 0, stream>>>(wo, wob, EMB * EMB);

  // qkv = x @ w_qkv^T   (M=8192, N=3072, K=1024)
  gemm_bf16_wmma<0><<<dim3(MT / 128, QKVW / 128), 256, 0, stream>>>(
      xb, wqkvb, (void*)qkvb, MT, QKVW, EMB);

  // flash attention
  flash_attn_bf16<<<dim3(TSEQ / 128, HEADS, BATCH), 256, 0, stream>>>(qkvb, yb);

  // out = y @ wo^T      (M=8192, N=1024, K=1024)
  gemm_bf16_wmma<1><<<dim3(MT / 128, EMB / 128), 256, 0, stream>>>(
      yb, wob, (void*)out, MT, EMB, EMB);
}
